// SelfAttentionBlock_60679297958311
// MI455X (gfx1250) — matrix operand order repaired and checked by reading, not compiled, not run on hardware
//
#include <hip/hip_runtime.h>
#include <hip/hip_bf16.h>

// ---------------------------------------------------------------------------
// MI455X (gfx1250) LoRA transformer block.
// Compute-bound (~275 GFLOP, working set < 192MB L2) -> everything GEMM-shaped
// goes through v_wmma_f32_16x16x32_bf16 (wave32). LoRA adapters are folded
// into the base weights once per call (W' = W + B@A). K-slab staging uses the
// CDNA5 data movers when the toolchain exposes them: TDM tensor_load_to_lds
// (TENSORcnt) for the GEMM tiles, global_load_async_to_lds_b128 (ASYNCcnt)
// otherwise / for attention tiles; guarded fallbacks keep the compile clean.
// ---------------------------------------------------------------------------

#define DIMC   1024
#define HEADS  16
#define HDIM   64
#define DFFC   4096
#define BB     4
#define TT     2048
#define NTOK   (BB*TT)   // 8192

typedef __attribute__((ext_vector_type(16))) __bf16 v16bf;
typedef __attribute__((ext_vector_type(8)))  float  v8f;
typedef __attribute__((ext_vector_type(4)))  unsigned int u32x4;
typedef __attribute__((ext_vector_type(8)))  int  i32x8;
typedef __attribute__((ext_vector_type(4)))  int  i32x4;

typedef __attribute__((address_space(3))) void l_void;    // LDS
typedef __attribute__((address_space(1))) i32x4 i32x4_as1;// <4 x i32> global
typedef __attribute__((address_space(3))) i32x4 i32x4_as3;// <4 x i32> LDS

#if defined(__HIP_DEVICE_COMPILE__) && __has_builtin(__builtin_amdgcn_tensor_load_to_lds)
#define HAVE_TDM 1
#else
#define HAVE_TDM 0
#endif
#if defined(__HIP_DEVICE_COMPILE__) && __has_builtin(__builtin_amdgcn_global_load_async_to_lds_b128)
#define HAVE_ASYNC 1
#else
#define HAVE_ASYNC 0
#endif

union Frag { u32x4 u[2]; v16bf v; };

__device__ inline unsigned short f2bf(float f) {
    unsigned int u = __float_as_uint(f);
    u += 0x7fffu + ((u >> 16) & 1u);           // round-to-nearest-even
    return (unsigned short)(u >> 16);
}

__device__ inline v8f v8f_zero() {
    v8f z;
    #pragma unroll
    for (int i = 0; i < 8; i++) z[i] = 0.f;
    return z;
}

__device__ inline v8f wmma_bf16(const Frag& a, const Frag& b, v8f c) {
    return __builtin_amdgcn_wmma_f32_16x16x32_bf16(
        false, a.v, false, b.v, (short)0, c, false, false);
}

__device__ inline unsigned lds_addr_of(const void* p) {
    // AS3 pointers are 32-bit LDS byte offsets.
    return (unsigned)(size_t)(l_void*)p;
}

#if HAVE_ASYNC
__device__ inline void async_b128(const void* gsrc, void* ldst) {
    __builtin_amdgcn_global_load_async_to_lds_b128(
        (i32x4_as1*)gsrc, (i32x4_as3*)ldst, 0, 0);
}
__device__ inline void wait_asynccnt0() {
#if __has_builtin(__builtin_amdgcn_s_wait_asynccnt)
    __builtin_amdgcn_s_wait_asynccnt(0);
#else
    asm volatile("s_wait_asynccnt 0" ::: "memory");
#endif
}
#endif

#if HAVE_TDM
// TDM: DMA one 128-row x 64-byte tile (row stride K bf16) into LDS with the
// DMA engine inserting +16B padding after every 64B row (LDS stride 80B).
// D# packing per CDNA5 ISA ch.8: group0 {count=1, lds_addr, global_addr,
// type=2}; group1 {data_size=8B, pad 64B->+16B, dims/tiles/strides}.
__device__ inline void tdm_load_tile_128x64(unsigned lds_byte_addr,
                                            const unsigned short* gsrc, int K) {
    unsigned long long ga = (unsigned long long)(size_t)gsrc;
    u32x4 g0;
    g0[0] = 1u;                                               // count=1 (user)
    g0[1] = lds_byte_addr;                                    // lds_addr
    g0[2] = (unsigned)ga;                                     // global_addr lo
    g0[3] = ((unsigned)(ga >> 32) & 0x01FFFFFFu) | (2u << 30);// addr hi | type=2
    i32x8 g1;
    g1[0] = (3 << 16) | (1 << 20) | (3 << 22) | (3 << 25);    // 8B units, pad_en,
                                                              // interval=16dw, amt=4dw
    g1[1] = (int)(8u << 16);      // abar=0 | tensor_dim0[15:0]=8 units (64B row)
    g1[2] = (int)(128u << 16);    // tensor_dim0 hi=0 | tensor_dim1[15:0]=128
    g1[3] = (int)(8u << 16);      // tensor_dim1 hi=0 | tile_dim0=8
    g1[4] = 128;                  // tile_dim1=128, tile_dim2=0
    g1[5] = K / 4;                // tensor_dim0_stride (8B units) = K*2B/8
    g1[6] = 0;                    // stride0 hi | tensor_dim1_stride lo
    g1[7] = 0;                    // tensor_dim1_stride hi
    i32x4 z4 = {0, 0, 0, 0};
#if __clang_major__ >= 23
    i32x8 z8 = {0, 0, 0, 0, 0, 0, 0, 0};
    __builtin_amdgcn_tensor_load_to_lds(g0, g1, z4, z4, z8, 0);
#else
    __builtin_amdgcn_tensor_load_to_lds(g0, g1, z4, z4, 0);
#endif
}
#endif

// ---------------------------------------------------------------------------
// 1) Fold LoRA: out_bf16[o][i] = bf16( W[o][i] + sum_r Bm[o][r]*A[r][i] )
// ---------------------------------------------------------------------------
__global__ __launch_bounds__(256)
void fuse_lora_kernel(const float* __restrict__ W, const float* __restrict__ A,
                      const float* __restrict__ Bm, unsigned short* __restrict__ out,
                      int dout, int din) {
    int idx = blockIdx.x * 256 + threadIdx.x;
    if (idx >= dout * din) return;
    int o = idx / din, i = idx - o * din;
    float acc = W[idx];
    #pragma unroll
    for (int r = 0; r < 16; r++) acc += Bm[o * 16 + r] * A[r * din + i];
    out[idx] = f2bf(acc);
}

// ---------------------------------------------------------------------------
// 2) LayerNorm (fp32 in) -> bf16 out.  One row (D=1024) per 256-thread block.
// ---------------------------------------------------------------------------
__device__ inline float block_sum256(float v, float* sm) {
    #pragma unroll
    for (int off = 16; off > 0; off >>= 1) v += __shfl_xor(v, off, 32);
    int wave = threadIdx.x >> 5, lane = threadIdx.x & 31;
    __syncthreads();
    if (lane == 0) sm[wave] = v;
    __syncthreads();
    float t = (lane < 8) ? sm[lane] : 0.f;
    #pragma unroll
    for (int off = 4; off > 0; off >>= 1) t += __shfl_xor(t, off, 32);
    __syncthreads();
    if (threadIdx.x == 0) sm[0] = t;
    __syncthreads();
    return sm[0];
}

__global__ __launch_bounds__(256)
void ln_bf16_kernel(const float* __restrict__ x, const float* __restrict__ g,
                    const float* __restrict__ b, unsigned short* __restrict__ out) {
    __shared__ float sm[8];
    const size_t row = blockIdx.x;
    const float* xr = x + row * DIMC;
    float vals[4], s = 0.f;
    #pragma unroll
    for (int i = 0; i < 4; i++) { vals[i] = xr[threadIdx.x + i * 256]; s += vals[i]; }
    float mu = block_sum256(s, sm) * (1.f / DIMC);
    float vs = 0.f;
    #pragma unroll
    for (int i = 0; i < 4; i++) { float d = vals[i] - mu; vs += d * d; }
    float var = block_sum256(vs, sm) * (1.f / DIMC);
    float inv = rsqrtf(var + 1e-5f);
    #pragma unroll
    for (int i = 0; i < 4; i++) {
        int c = threadIdx.x + i * 256;
        out[row * DIMC + c] = f2bf((vals[i] - mu) * inv * g[c] + b[c]);
    }
}

// ---------------------------------------------------------------------------
// 3) WMMA GEMM: out[M,N] = X[M,K](bf16) @ W[N,K](bf16)^T + bias, epilogue:
//    EPI=0: store bf16   EPI=1: fp32 out = res + acc   EPI=2: exact GELU->bf16
//    Workgroup tile 128x128, 8 waves of 32x64, K staged 32-wide via LDS
//    (TDM -> async-to-LDS -> manual, by toolchain capability).
// ---------------------------------------------------------------------------
template <int EPI>
__global__ __launch_bounds__(256)
void gemm_wmma_kernel(const unsigned short* __restrict__ X,
                      const unsigned short* __restrict__ W,
                      const float* __restrict__ bias,
                      const float* __restrict__ res,
                      void* __restrict__ out, int M, int N, int K) {
    __shared__ __align__(16) unsigned short Xs[128][40];
    __shared__ __align__(16) unsigned short Ws[128][40];
    const int tid = threadIdx.x;
    const int lane = tid & 31, wave = tid >> 5;
    const int half = lane >> 4, l15 = lane & 15;
    const int bm = blockIdx.y * 128, bn = blockIdx.x * 128;
    const int wm = (wave >> 1) * 32;   // 0,32,64,96
    const int wn = (wave & 1) * 64;    // 0,64

    v8f acc[2][4];
    #pragma unroll
    for (int i = 0; i < 2; i++)
        #pragma unroll
        for (int j = 0; j < 4; j++) acc[i][j] = v8f_zero();

    for (int kc = 0; kc < K; kc += 32) {
        __syncthreads();  // previous iteration's LDS reads done
#if HAVE_TDM
        if (wave == 0) {
            tdm_load_tile_128x64(lds_addr_of(&Xs[0][0]), &X[(size_t)bm * K + kc], K);
            tdm_load_tile_128x64(lds_addr_of(&Ws[0][0]), &W[(size_t)bn * K + kc], K);
            __builtin_amdgcn_s_wait_tensorcnt(0);
        }
#elif HAVE_ASYNC
        #pragma unroll
        for (int s = 0; s < 2; s++) {
            int idx = tid + s * 256;
            int row = idx >> 2, ch = (idx & 3) * 8;
            async_b128(&X[(size_t)(bm + row) * K + kc + ch], &Xs[row][ch]);
            async_b128(&W[(size_t)(bn + row) * K + kc + ch], &Ws[row][ch]);
        }
        wait_asynccnt0();
#else
        #pragma unroll
        for (int s = 0; s < 2; s++) {
            int idx = tid + s * 256;
            int row = idx >> 2, ch = (idx & 3) * 8;
            *(u32x4*)&Xs[row][ch] = *(const u32x4*)&X[(size_t)(bm + row) * K + kc + ch];
            *(u32x4*)&Ws[row][ch] = *(const u32x4*)&W[(size_t)(bn + row) * K + kc + ch];
            if (kc + 32 < K) {   // global_prefetch_b8 for next K-slab
                __builtin_prefetch(&X[(size_t)(bm + row) * K + kc + 32 + ch], 0, 0);
                __builtin_prefetch(&W[(size_t)(bn + row) * K + kc + 32 + ch], 0, 0);
            }
        }
#endif
        __syncthreads();
        Frag a[2], bfr[4];
        #pragma unroll
        for (int i = 0; i < 2; i++) {   // A: rows within lane, K contiguous by half
            const unsigned short* p = &Xs[wm + i * 16 + l15][half * 16];
            a[i].u[0] = *(const u32x4*)p;
            a[i].u[1] = *(const u32x4*)(p + 8);
        }
        #pragma unroll
        for (int j = 0; j < 4; j++) {   // B: col per lane, K contiguous by half
            const unsigned short* p = &Ws[wn + j * 16 + l15][half * 16];
            bfr[j].u[0] = *(const u32x4*)p;
            bfr[j].u[1] = *(const u32x4*)(p + 8);
        }
        #pragma unroll
        for (int i = 0; i < 2; i++)
            #pragma unroll
            for (int j = 0; j < 4; j++)
                acc[i][j] = wmma_bf16(a[i], bfr[j], acc[i][j]);
    }

    #pragma unroll
    for (int i = 0; i < 2; i++) {
        int rbase = bm + wm + i * 16 + half * 8;
        #pragma unroll
        for (int j = 0; j < 4; j++) {
            int col = bn + wn + j * 16 + l15;
            float bv = bias ? bias[col] : 0.f;
            #pragma unroll
            for (int v = 0; v < 8; v++) {
                float val = acc[i][j][v] + bv;
                size_t off = (size_t)(rbase + v) * N + col;
                if (EPI == 0) {
                    ((unsigned short*)out)[off] = f2bf(val);
                } else if (EPI == 1) {
                    ((float*)out)[off] = res[off] + val;
                } else {
                    float gl = 0.5f * val * (1.f + erff(val * 0.70710678118f));
                    ((unsigned short*)out)[off] = f2bf(gl);
                }
            }
        }
    }
}

// ---------------------------------------------------------------------------
// 4) Attention: per (b,h,128 q-rows) workgroup, flash-style online softmax.
//    8 waves x 16 q-rows. K tile (async-to-LDS when available) + transposed
//    V tile staged in LDS so every WMMA fragment is a contiguous b128 LDS
//    read. P relayout C->A via per-wave LDS patch.
// ---------------------------------------------------------------------------
__global__ __launch_bounds__(256)
void attention_kernel(const unsigned short* __restrict__ Q,
                      const unsigned short* __restrict__ Kb,
                      const unsigned short* __restrict__ Vb,
                      unsigned short* __restrict__ Out) {
    __shared__ __align__(16) unsigned short Ks[32][72];     // 32 keys x 64 d
    __shared__ __align__(16) unsigned short Vts[64][40];    // 64 d x 32 keys
    __shared__ __align__(16) unsigned short Plds[8][16][40];// per-wave P tile

    const int tid = threadIdx.x, lane = tid & 31, wave = tid >> 5;
    const int half = lane >> 4, l15 = lane & 15;
    const int qblk = blockIdx.x * 128;
    const int h = blockIdx.y, bidx = blockIdx.z;
    const size_t bhoff = (size_t)bidx * TT * DIMC + (size_t)h * HDIM;

    // Q fragments (A layout), held in registers for the whole key loop
    const int qrow = qblk + wave * 16 + l15;
    Frag qf[2];
    const unsigned short* qp = Q + bhoff + (size_t)qrow * DIMC;
    #pragma unroll
    for (int f = 0; f < 2; f++) {
        qf[f].u[0] = *(const u32x4*)(qp + f * 32 + half * 16);
        qf[f].u[1] = *(const u32x4*)(qp + f * 32 + half * 16 + 8);
    }

    v8f acc[4];
    #pragma unroll
    for (int j = 0; j < 4; j++) acc[j] = v8f_zero();
    float mrow[8], lrow[8];
    #pragma unroll
    for (int v = 0; v < 8; v++) { mrow[v] = -1e30f; lrow[v] = 0.f; }

    for (int kb = 0; kb < TT; kb += 32) {
        __syncthreads();
        {   // stage K tile (natural) + V tile (transposed)
            int key = tid >> 3, ch = (tid & 7) * 8;
            const unsigned short* kp = Kb + bhoff + (size_t)(kb + key) * DIMC + ch;
#if HAVE_ASYNC
            async_b128(kp, &Ks[key][ch]);
#else
            *(u32x4*)&Ks[key][ch] = *(const u32x4*)kp;
#endif
            u32x4 vv = *(const u32x4*)(Vb + bhoff + (size_t)(kb + key) * DIMC + ch);
            #pragma unroll
            for (int e = 0; e < 8; e++) {
                unsigned int w = vv[e >> 1];
                Vts[ch + e][key] = (unsigned short)((e & 1) ? (w >> 16) : (w & 0xffffu));
            }
        }
#if HAVE_ASYNC
        wait_asynccnt0();
#endif
        __syncthreads();

        // S = Q @ K^T for 32 keys (two 16x16 C tiles, 2 WMMAs each over d=64)
        v8f s0 = v8f_zero(), s1 = v8f_zero();
        {
            const unsigned short* p = &Ks[l15][0];
            Frag b0, b1;
            b0.u[0] = *(const u32x4*)(p + half * 16);
            b0.u[1] = *(const u32x4*)(p + half * 16 + 8);
            b1.u[0] = *(const u32x4*)(p + 32 + half * 16);
            b1.u[1] = *(const u32x4*)(p + 32 + half * 16 + 8);
            s0 = wmma_bf16(qf[0], b0, s0);
            s0 = wmma_bf16(qf[1], b1, s0);
        }
        {
            const unsigned short* p = &Ks[16 + l15][0];
            Frag b0, b1;
            b0.u[0] = *(const u32x4*)(p + half * 16);
            b0.u[1] = *(const u32x4*)(p + half * 16 + 8);
            b1.u[0] = *(const u32x4*)(p + 32 + half * 16);
            b1.u[1] = *(const u32x4*)(p + 32 + half * 16 + 8);
            s1 = wmma_bf16(qf[0], b0, s1);
            s1 = wmma_bf16(qf[1], b1, s1);
        }

        // online softmax update (row = v + 8*half, 16 lanes per row-half)
        float p0[8], p1[8];
        #pragma unroll
        for (int v = 0; v < 8; v++) {
            float a0 = s0[v] * 0.125f, a1 = s1[v] * 0.125f;  // 1/sqrt(64)
            float t = fmaxf(a0, a1);
            #pragma unroll
            for (int off = 8; off > 0; off >>= 1) t = fmaxf(t, __shfl_xor(t, off, 32));
            float mn = fmaxf(mrow[v], t);
            float alpha = expf(mrow[v] - mn);
            mrow[v] = mn;
            p0[v] = expf(a0 - mn);
            p1[v] = expf(a1 - mn);
            float r = p0[v] + p1[v];
            #pragma unroll
            for (int off = 8; off > 0; off >>= 1) r += __shfl_xor(r, off, 32);
            lrow[v] = lrow[v] * alpha + r;
            #pragma unroll
            for (int j = 0; j < 4; j++) acc[j][v] *= alpha;
        }

        // P tile (C layout) -> LDS -> A fragment layout
        #pragma unroll
        for (int v = 0; v < 8; v++) {
            Plds[wave][v + 8 * half][l15]      = f2bf(p0[v]);
            Plds[wave][v + 8 * half][16 + l15] = f2bf(p1[v]);
        }
        __syncthreads();
        Frag pf;
        {
            const unsigned short* pp = &Plds[wave][l15][0];
            pf.u[0] = *(const u32x4*)(pp + half * 16);
            pf.u[1] = *(const u32x4*)(pp + half * 16 + 8);
        }
        #pragma unroll
        for (int j = 0; j < 4; j++) {   // O += P @ V  (B from transposed V)
            const unsigned short* vp = &Vts[j * 16 + l15][0];
            Frag vf;
            vf.u[0] = *(const u32x4*)(vp + half * 16);
            vf.u[1] = *(const u32x4*)(vp + half * 16 + 8);
            acc[j] = wmma_bf16(pf, vf, acc[j]);
        }
    }

    #pragma unroll
    for (int j = 0; j < 4; j++)
        #pragma unroll
        for (int v = 0; v < 8; v++) {
            int row = qblk + wave * 16 + v + 8 * half;
            float o = acc[j][v] / lrow[v];
            Out[bhoff + (size_t)row * DIMC + j * 16 + l15] = f2bf(o);
        }
}

// ---------------------------------------------------------------------------
// Launch
// ---------------------------------------------------------------------------
extern "C" void kernel_launch(void* const* d_in, const int* in_sizes, int n_in,
                              void* d_out, int out_size, void* d_ws, size_t ws_size,
                              hipStream_t stream) {
    const float* x     = (const float*)d_in[0];
    const float* ln1g  = (const float*)d_in[1];
    const float* ln1b  = (const float*)d_in[2];
    const float* ln2g  = (const float*)d_in[3];
    const float* ln2b  = (const float*)d_in[4];
    const float *q_W = (const float*)d_in[5],  *q_b = (const float*)d_in[6],
                *q_A = (const float*)d_in[7],  *q_Bm = (const float*)d_in[8];
    const float *k_W = (const float*)d_in[9],  *k_b = (const float*)d_in[10],
                *k_A = (const float*)d_in[11], *k_Bm = (const float*)d_in[12];
    const float *v_W = (const float*)d_in[13], *v_b = (const float*)d_in[14],
                *v_A = (const float*)d_in[15], *v_Bm = (const float*)d_in[16];
    const float *o_W = (const float*)d_in[17], *o_b = (const float*)d_in[18],
                *o_A = (const float*)d_in[19], *o_Bm = (const float*)d_in[20];
    const float *f1_W = (const float*)d_in[21], *f1_b = (const float*)d_in[22],
                *f1_A = (const float*)d_in[23], *f1_Bm = (const float*)d_in[24];
    const float *f2_W = (const float*)d_in[25], *f2_b = (const float*)d_in[26],
                *f2_A = (const float*)d_in[27], *f2_Bm = (const float*)d_in[28];

    size_t off = 0;
    auto alloc = [&](size_t bytes) -> void* {
        void* p = (char*)d_ws + off;
        off = (off + bytes + 255) & ~(size_t)255;
        return p;
    };
    unsigned short* wq  = (unsigned short*)alloc((size_t)DIMC * DIMC * 2);
    unsigned short* wk  = (unsigned short*)alloc((size_t)DIMC * DIMC * 2);
    unsigned short* wv  = (unsigned short*)alloc((size_t)DIMC * DIMC * 2);
    unsigned short* wo  = (unsigned short*)alloc((size_t)DIMC * DIMC * 2);
    unsigned short* wf1 = (unsigned short*)alloc((size_t)DFFC * DIMC * 2);
    unsigned short* wf2 = (unsigned short*)alloc((size_t)DIMC * DFFC * 2);
    unsigned short* h1  = (unsigned short*)alloc((size_t)NTOK * DIMC * 2);
    unsigned short* qbf = (unsigned short*)alloc((size_t)NTOK * DIMC * 2);
    unsigned short* kbf = (unsigned short*)alloc((size_t)NTOK * DIMC * 2);
    unsigned short* vbf = (unsigned short*)alloc((size_t)NTOK * DIMC * 2);
    unsigned short* atn = (unsigned short*)alloc((size_t)NTOK * DIMC * 2);
    float*          x1  = (float*)alloc((size_t)NTOK * DIMC * 4);
    unsigned short* h2  = (unsigned short*)alloc((size_t)NTOK * DIMC * 2);
    unsigned short* ff1o= (unsigned short*)alloc((size_t)NTOK * DFFC * 2);

    // fold LoRA into base weights (bf16)
    fuse_lora_kernel<<<(DIMC*DIMC+255)/256, 256, 0, stream>>>(q_W, q_A, q_Bm, wq, DIMC, DIMC);
    fuse_lora_kernel<<<(DIMC*DIMC+255)/256, 256, 0, stream>>>(k_W, k_A, k_Bm, wk, DIMC, DIMC);
    fuse_lora_kernel<<<(DIMC*DIMC+255)/256, 256, 0, stream>>>(v_W, v_A, v_Bm, wv, DIMC, DIMC);
    fuse_lora_kernel<<<(DIMC*DIMC+255)/256, 256, 0, stream>>>(o_W, o_A, o_Bm, wo, DIMC, DIMC);
    fuse_lora_kernel<<<(DFFC*DIMC+255)/256, 256, 0, stream>>>(f1_W, f1_A, f1_Bm, wf1, DFFC, DIMC);
    fuse_lora_kernel<<<(DIMC*DFFC+255)/256, 256, 0, stream>>>(f2_W, f2_A, f2_Bm, wf2, DIMC, DFFC);

    // LN1 -> h1 (bf16)
    ln_bf16_kernel<<<NTOK, 256, 0, stream>>>(x, ln1g, ln1b, h1);

    // Q/K/V projections (bf16 out for attention)
    dim3 g1(DIMC/128, NTOK/128);
    gemm_wmma_kernel<0><<<g1, 256, 0, stream>>>(h1, wq, q_b, nullptr, qbf, NTOK, DIMC, DIMC);
    gemm_wmma_kernel<0><<<g1, 256, 0, stream>>>(h1, wk, k_b, nullptr, kbf, NTOK, DIMC, DIMC);
    gemm_wmma_kernel<0><<<g1, 256, 0, stream>>>(h1, wv, v_b, nullptr, vbf, NTOK, DIMC, DIMC);

    // attention
    attention_kernel<<<dim3(TT/128, HEADS, BB), 256, 0, stream>>>(qbf, kbf, vbf, atn);

    // O projection + residual (fp32 x1)
    gemm_wmma_kernel<1><<<g1, 256, 0, stream>>>(atn, wo, o_b, x, x1, NTOK, DIMC, DIMC);

    // LN2 -> h2 (bf16)
    ln_bf16_kernel<<<NTOK, 256, 0, stream>>>(x1, ln2g, ln2b, h2);

    // FF1 + exact GELU (bf16), FF2 + residual (fp32 final output)
    gemm_wmma_kernel<2><<<dim3(DFFC/128, NTOK/128), 256, 0, stream>>>(h2, wf1, f1_b, nullptr, ff1o, NTOK, DFFC, DIMC);
    gemm_wmma_kernel<1><<<g1, 256, 0, stream>>>(ff1o, wf2, f2_b, x1, (float*)d_out, NTOK, DIMC, DFFC);
}
